// AgentModule_86208583565990
// MI455X (gfx1250) — compile-verified, matrix operand order checked by reading, not run on hardware
//
#include <hip/hip_runtime.h>
#include <math.h>

// ---------------------------------------------------------------------------
// Problem constants: B=512, A=4, E=10, H=256, OBS=2, PHY=3, G=3, MV=2, T=10
// ---------------------------------------------------------------------------
#define HH   256
#define G3   768            // 3*H
#define NBA  2048           // B*A
#define NBAE 20480          // B*A*E
#define NE   10
#define NT   10
#define STEPSZ 0.05f

typedef __attribute__((ext_vector_type(16))) __bf16 v16bf;
typedef __attribute__((ext_vector_type(8)))  float  v8f;

// ---------------------------------------------------------------------------
// WMMA fragment helpers (CDNA5 ISA 7.12.2, 16-bit layouts, wave32)
// ---------------------------------------------------------------------------
__device__ __forceinline__ int frag_kk(int v, int half) {
    return (v < 4) ? (half * 8 + 2 * v) : (16 + half * 8 + 2 * (v - 4));
}

__device__ __forceinline__ v16bf load_fragA_lds(const __bf16* tile, int ldt,
                                                int lane, int kbase) {
    int m = lane & 15, half = lane >> 4;
    v16bf a;
#pragma unroll
    for (int v = 0; v < 8; ++v) {
        int kk = frag_kk(v, half);
        a[2 * v]     = tile[m * ldt + kbase + kk];
        a[2 * v + 1] = tile[m * ldt + kbase + kk + 1];
    }
    return a;
}

// B fragment from an LDS-staged packed n-tile (16 contiguous bf16 per lane).
__device__ __forceinline__ v16bf load_fragB_lds(const __bf16* sec, int ks, int lane) {
    return *(const v16bf*)(sec + ((ks * 32 + lane) << 4));
}

__device__ __forceinline__ v8f wmma_bf16(v16bf a, v16bf b, v8f c) {
    return __builtin_amdgcn_wmma_f32_16x16x32_bf16(
        false, a, false, b, (short)0, c, false, false);
}

// Cooperative copy of one packed n-tile (4096 bf16 = 8KB, contiguous) to LDS.
__device__ __forceinline__ void stage_ntile(const __bf16* pk, int nt,
                                            __bf16* dst, int t) {
    const uint4* s = (const uint4*)(pk + ((size_t)nt << 12));
    uint4* d = (uint4*)dst;        // 512 uint4 / 256 threads = 2 each
    d[t]       = s[t];
    d[t + 256] = s[t + 256];
}

__device__ __forceinline__ float sigm(float x) { return 1.f / (1.f + __expf(-x)); }
__device__ __forceinline__ float eluf(float x) { return x > 0.f ? x : (__expf(x) - 1.f); }

// ---------------------------------------------------------------------------
// K0: pack f32 W[N x K] (row stride ldw) into bf16 B-fragment layout.
//     Layout is contiguous per n-tile: [nt][ks][lane][16]
// ---------------------------------------------------------------------------
__global__ void k_pack_w(const float* __restrict__ W, int N, int K, int ldw,
                         __bf16* __restrict__ out) {
    int idx = blockIdx.x * 256 + threadIdx.x;
    if (idx >= N * K) return;
    int j    = idx & 15;
    int lane = (idx >> 4) & 31;
    int blk  = idx >> 9;                 // nt*KS + ks
    int KS   = K >> 5;
    int nt   = blk / KS, ks = blk % KS;
    int n    = nt * 16 + (lane & 15);
    int half = lane >> 4;
    int v = j >> 1, lo = j & 1;
    int k = ks * 32 + frag_kk(v, half) + lo;
    out[idx] = (__bf16)W[(size_t)n * ldw + k];
}

// ---------------------------------------------------------------------------
// K1a: gi_p[20480,768] = x_p @ W_ih_p^T + b_ih_p   (K=5, step-invariant)
// ---------------------------------------------------------------------------
__global__ void k_gi_p(const float* __restrict__ obs, const float* __restrict__ phys,
                       const float* __restrict__ Wih, const float* __restrict__ bih,
                       float* __restrict__ gi) {
    size_t idx = (size_t)blockIdx.x * 256 + threadIdx.x;
    if (idx >= (size_t)NBAE * G3) return;
    int n = (int)(idx % G3);
    int r = (int)(idx / G3);             // ((b*A+a)*E + e)
    int e  = r % NE;
    int ba = r / NE;
    int b  = ba >> 2;
    const float* ob = obs + (size_t)r * 2;
    const float* ph = phys + ((size_t)b * NE + e) * 3;
    const float* w  = Wih + (size_t)n * 5;
    gi[idx] = bih[n] + ob[0]*w[0] + ob[1]*w[1] + ph[0]*w[2] + ph[1]*w[3] + ph[2]*w[4];
}

// K1b: gia_goal[2048,768] = goals @ W_ih_a[:,256:259]^T + b_ih_a
__global__ void k_gia_goal(const float* __restrict__ goals,
                           const float* __restrict__ Wih_a,
                           const float* __restrict__ bih_a,
                           float* __restrict__ gia) {
    size_t idx = (size_t)blockIdx.x * 256 + threadIdx.x;
    if (idx >= (size_t)NBA * G3) return;
    int n = (int)(idx % G3);
    int r = (int)(idx / G3);
    const float* g = goals + (size_t)r * 3;
    const float* w = Wih_a + (size_t)n * 259 + 256;
    gia[idx] = bih_a[n] + g[0]*w[0] + g[1]*w[1] + g[2]*w[2];
}

// ---------------------------------------------------------------------------
// K2: physical GRU step. Block = 128 rows, 8 waves; wave owns one 16-row
//     M-tile (A frags register-resident: 64 VGPRs, no spill). B n-tiles are
//     cooperatively staged in LDS -> weights read from L2 ONCE per block.
//     Gates fused into the column loop (r,z,n column blocks together), so no
//     gh LDS array is needed at all.
// ---------------------------------------------------------------------------
__global__ void k_phys_step(const float* __restrict__ gi_p,
                            const float* __restrict__ bhh,
                            const float* __restrict__ bfc,
                            const __bf16* __restrict__ Whh_pk,
                            const __bf16* __restrict__ Wfc_pk,
                            float* __restrict__ mp,
                            float* __restrict__ proc) {
    const int row0 = blockIdx.x * 128;
    const int t = threadIdx.x, lane = t & 31, wave = t >> 5;
    const int n = lane & 15, half = lane >> 4;

    __shared__ __bf16 sA[128][HH + 8];              // state tile -> later h tile
    __shared__ __align__(16) __bf16 sB[3 * 4096];   // 3 staged B n-tiles (24KB)

    for (int i = t; i < 128 * HH; i += 256) {
        int m = i >> 8, k = i & 255;
        sA[m][k] = (__bf16)mp[(size_t)(row0 + m) * HH + k];
    }
    __syncthreads();

    // Hoist this wave's A fragments (its own 16 rows) into registers.
    v16bf aF[8];
#pragma unroll
    for (int ks = 0; ks < 8; ++ks)
        aF[ks] = load_fragA_lds(&sA[wave * 16][0], HH + 8, lane, ks * 32);

    // Fused GEMM + GRU gates, one 16-column block (r,z,n triple) at a time.
    for (int nt2 = 0; nt2 < 16; ++nt2) {
        __syncthreads();                            // prior sB consumers done
        stage_ntile(Whh_pk, nt2,      sB,        t);
        stage_ntile(Whh_pk, nt2 + 16, sB + 4096, t);
        stage_ntile(Whh_pk, nt2 + 32, sB + 8192, t);
        __syncthreads();

        v8f ar = {}, az = {}, an = {};
#pragma unroll
        for (int ks = 0; ks < 8; ++ks) {
            v16bf br = load_fragB_lds(sB,        ks, lane);
            v16bf bz = load_fragB_lds(sB + 4096, ks, lane);
            v16bf bn = load_fragB_lds(sB + 8192, ks, lane);
            ar = wmma_bf16(aF[ks], br, ar);
            az = wmma_bf16(aF[ks], bz, az);
            an = wmma_bf16(aF[ks], bn, an);
        }
        int col = nt2 * 16 + n;
        float bhr = bhh[col], bhz = bhh[HH + col], bhn = bhh[2 * HH + col];
#pragma unroll
        for (int v = 0; v < 8; ++v) {
            int lrow = wave * 16 + v + 8 * half;
            int grow = row0 + lrow;
            const float* gp = gi_p + (size_t)grow * G3;
            float r  = sigm(gp[col] + ar[v] + bhr);
            float z  = sigm(gp[HH + col] + az[v] + bhz);
            float nn = tanhf(gp[2 * HH + col] + r * (an[v] + bhn));
            size_t ix = (size_t)grow * HH + col;
            float hold = mp[ix];
            float h = (1.f - z) * nn + z * hold;
            mp[ix] = h;
            sA[lrow][col] = (__bf16)h;              // own rows only: safe
        }
    }
    __syncthreads();

    // Re-hoist A fragments = new h.
#pragma unroll
    for (int ks = 0; ks < 8; ++ks)
        aF[ks] = load_fragA_lds(&sA[wave * 16][0], HH + 8, lane, ks * 32);

    // proc = elu(h @ Wfc^T + bfc)
    for (int nt2 = 0; nt2 < 16; ++nt2) {
        __syncthreads();
        stage_ntile(Wfc_pk, nt2, sB, t);
        __syncthreads();
        v8f acc = {};
#pragma unroll
        for (int ks = 0; ks < 8; ++ks)
            acc = wmma_bf16(aF[ks], load_fragB_lds(sB, ks, lane), acc);
        float bb = bfc[nt2 * 16 + n];
#pragma unroll
        for (int v = 0; v < 8; ++v)
            proc[(size_t)(row0 + wave * 16 + v + 8 * half) * HH + nt2 * 16 + n] =
                eluf(acc[v] + bb);
    }
}

// ---------------------------------------------------------------------------
// K3: feat[ba][j] = max_e proc[ba*10+e][j]
// ---------------------------------------------------------------------------
__global__ void k_pool(const float* __restrict__ proc, float* __restrict__ feat) {
    int ba = blockIdx.x, j = threadIdx.x;
    float m = -3.4e38f;
    const float* p = proc + (size_t)ba * NE * HH + j;
#pragma unroll
    for (int e = 0; e < NE; ++e) m = fmaxf(m, p[(size_t)e * HH]);
    feat[(size_t)ba * HH + j] = m;
}

// ---------------------------------------------------------------------------
// K4: fused action chain. Block = 128 rows, 8 waves, wave owns one M-tile.
//     r/z sections accumulate gi+gh in one chain (exact); n kept separate.
//     A fragments gathered in-loop (low register pressure, no spills).
// ---------------------------------------------------------------------------
__global__ void k_act_step(const float* __restrict__ feat,
                           const float* __restrict__ gia_goal,
                           const float* __restrict__ bhh,
                           const float* __restrict__ bfc,
                           const float* __restrict__ bm1,
                           const float* __restrict__ Wm2,
                           const float* __restrict__ bm2,
                           const __bf16* __restrict__ Wihf_pk,
                           const __bf16* __restrict__ Whh_pk,
                           const __bf16* __restrict__ Wfc_pk,
                           const __bf16* __restrict__ Wm1_pk,
                           float* __restrict__ ma,
                           float* __restrict__ out_t) {
    const int row0 = blockIdx.x * 128;
    const int t = threadIdx.x, lane = t & 31, wave = t >> 5;
    const int n = lane & 15, half = lane >> 4;

    __shared__ __bf16 sF[128][HH + 8];              // feat -> later m1 (bf16)
    __shared__ __bf16 sM[128][HH + 8];              // ma   -> later proc_a
    __shared__ __bf16 sH[128][HH + 8];              // new h
    __shared__ __align__(16) __bf16 sB[6 * 4096];   // staged B n-tiles (48KB)

    for (int i = t; i < 128 * HH; i += 256) {
        int m = i >> 8, k = i & 255;
        sF[m][k] = (__bf16)feat[(size_t)(row0 + m) * HH + k];
        sM[m][k] = (__bf16)ma[(size_t)(row0 + m) * HH + k];
    }

    // GRU GEMMs + gates, one 16-column block at a time.
    for (int nt2 = 0; nt2 < 16; ++nt2) {
        __syncthreads();
        stage_ntile(Wihf_pk, nt2,      sB,            t);
        stage_ntile(Whh_pk,  nt2,      sB + 1 * 4096, t);
        stage_ntile(Wihf_pk, nt2 + 16, sB + 2 * 4096, t);
        stage_ntile(Whh_pk,  nt2 + 16, sB + 3 * 4096, t);
        stage_ntile(Wihf_pk, nt2 + 32, sB + 4 * 4096, t);
        stage_ntile(Whh_pk,  nt2 + 32, sB + 5 * 4096, t);
        __syncthreads();

        v8f ar = {}, az = {}, ani = {}, anh = {};
#pragma unroll
        for (int ks = 0; ks < 8; ++ks) {
            v16bf af = load_fragA_lds(&sF[wave * 16][0], HH + 8, lane, ks * 32);
            v16bf am = load_fragA_lds(&sM[wave * 16][0], HH + 8, lane, ks * 32);
            ar  = wmma_bf16(af, load_fragB_lds(sB,            ks, lane), ar);
            ar  = wmma_bf16(am, load_fragB_lds(sB + 1 * 4096, ks, lane), ar);
            az  = wmma_bf16(af, load_fragB_lds(sB + 2 * 4096, ks, lane), az);
            az  = wmma_bf16(am, load_fragB_lds(sB + 3 * 4096, ks, lane), az);
            ani = wmma_bf16(af, load_fragB_lds(sB + 4 * 4096, ks, lane), ani);
            anh = wmma_bf16(am, load_fragB_lds(sB + 5 * 4096, ks, lane), anh);
        }
        int col = nt2 * 16 + n;
        float bhr = bhh[col], bhz = bhh[HH + col], bhn = bhh[2 * HH + col];
#pragma unroll
        for (int v = 0; v < 8; ++v) {
            int lrow = wave * 16 + v + 8 * half;
            int grow = row0 + lrow;
            const float* gg = gia_goal + (size_t)grow * G3;
            float r  = sigm(ar[v] + gg[col] + bhr);
            float z  = sigm(az[v] + gg[HH + col] + bhz);
            float nn = tanhf((ani[v] + gg[2 * HH + col]) + r * (anh[v] + bhn));
            size_t ix = (size_t)grow * HH + col;
            float hold = ma[ix];
            float h = (1.f - z) * nn + z * hold;
            ma[ix] = h;
            sH[lrow][col] = (__bf16)h;
        }
    }
    __syncthreads();

    // proc_a = elu(h @ Wfc_a^T + bfc) -> sM (ma dead)
    for (int nt2 = 0; nt2 < 16; ++nt2) {
        __syncthreads();
        stage_ntile(Wfc_pk, nt2, sB, t);
        __syncthreads();
        v8f acc = {};
#pragma unroll
        for (int ks = 0; ks < 8; ++ks) {
            v16bf a = load_fragA_lds(&sH[wave * 16][0], HH + 8, lane, ks * 32);
            acc = wmma_bf16(a, load_fragB_lds(sB, ks, lane), acc);
        }
        float bb = bfc[nt2 * 16 + n];
#pragma unroll
        for (int v = 0; v < 8; ++v)
            sM[wave * 16 + v + 8 * half][nt2 * 16 + n] = (__bf16)eluf(acc[v] + bb);
    }
    __syncthreads();

    // m1 = elu(proc_a @ Wm1^T + bm1) -> sF (feat dead)
    for (int nt2 = 0; nt2 < 16; ++nt2) {
        __syncthreads();
        stage_ntile(Wm1_pk, nt2, sB, t);
        __syncthreads();
        v8f acc = {};
#pragma unroll
        for (int ks = 0; ks < 8; ++ks) {
            v16bf a = load_fragA_lds(&sM[wave * 16][0], HH + 8, lane, ks * 32);
            acc = wmma_bf16(a, load_fragB_lds(sB, ks, lane), acc);
        }
        float bb = bm1[nt2 * 16 + n];
#pragma unroll
        for (int v = 0; v < 8; ++v)
            sF[wave * 16 + v + 8 * half][nt2 * 16 + n] = (__bf16)eluf(acc[v] + bb);
    }
    __syncthreads();

    // mv = tanh(m1 @ Wm2^T + bm2) * STEP  (N=2: one dot per thread)
    {
        int m = t >> 1, o = t & 1;
        float d = bm2[o];
        const float* w = Wm2 + (size_t)o * HH;
        for (int j = 0; j < HH; ++j) d += (float)sF[m][j] * w[j];
        out_t[(size_t)(row0 + m) * 2 + o] = tanhf(d) * STEPSZ;
    }
}

// ---------------------------------------------------------------------------
// Host side
// ---------------------------------------------------------------------------
static inline size_t align_up(size_t x) { return (x + 255) & ~(size_t)255; }

extern "C" void kernel_launch(void* const* d_in, const int* in_sizes, int n_in,
                              void* d_out, int out_size, void* d_ws, size_t ws_size,
                              hipStream_t stream) {
    const float* obs     = (const float*)d_in[0];   // [B,A,E,2]
    const float* phys    = (const float*)d_in[1];   // [B,E,3]
    const float* goals   = (const float*)d_in[2];   // [B,A,3]
    const float* mp_in   = (const float*)d_in[3];   // [B,A,E,H]
    const float* ma_in   = (const float*)d_in[4];   // [B,A,H]
    const float* W_ih_p  = (const float*)d_in[5];   // [768,5]
    const float* W_hh_p  = (const float*)d_in[6];   // [768,256]
    const float* b_ih_p  = (const float*)d_in[7];
    const float* b_hh_p  = (const float*)d_in[8];
    const float* W_fc_p  = (const float*)d_in[9];   // [256,256]
    const float* b_fc_p  = (const float*)d_in[10];
    const float* W_ih_a  = (const float*)d_in[11];  // [768,259]
    const float* W_hh_a  = (const float*)d_in[12];  // [768,256]
    const float* b_ih_a  = (const float*)d_in[13];
    const float* b_hh_a  = (const float*)d_in[14];
    const float* W_fc_a  = (const float*)d_in[15];  // [256,256]
    const float* b_fc_a  = (const float*)d_in[16];
    const float* W_m1    = (const float*)d_in[17];  // [256,256]
    const float* b_m1    = (const float*)d_in[18];
    const float* W_m2    = (const float*)d_in[19];  // [2,256]
    const float* b_m2    = (const float*)d_in[20];
    float* out = (float*)d_out;                     // [T,B,A,2]

    // Workspace carve
    char* p = (char*)d_ws;
    auto alloc = [&](size_t bytes) { char* r = p; p += align_up(bytes); return r; };
    float* mp       = (float*)alloc((size_t)NBAE * HH * 4);
    float* ma       = (float*)alloc((size_t)NBA * HH * 4);
    float* gi_p     = (float*)alloc((size_t)NBAE * G3 * 4);
    float* gia_goal = (float*)alloc((size_t)NBA * G3 * 4);
    float* proc     = (float*)alloc((size_t)NBAE * HH * 4);
    float* feat     = (float*)alloc((size_t)NBA * HH * 4);
    __bf16* whh_p_pk = (__bf16*)alloc((size_t)G3 * HH * 2);
    __bf16* wfc_p_pk = (__bf16*)alloc((size_t)HH * HH * 2);
    __bf16* wihf_pk  = (__bf16*)alloc((size_t)G3 * HH * 2);
    __bf16* whh_a_pk = (__bf16*)alloc((size_t)G3 * HH * 2);
    __bf16* wfc_a_pk = (__bf16*)alloc((size_t)HH * HH * 2);
    __bf16* wm1_pk   = (__bf16*)alloc((size_t)HH * HH * 2);
    (void)ws_size; (void)in_sizes; (void)n_in; (void)out_size;

    // State init (never mutate inputs)
    hipMemcpyAsync(mp, mp_in, (size_t)NBAE * HH * 4, hipMemcpyDeviceToDevice, stream);
    hipMemcpyAsync(ma, ma_in, (size_t)NBA * HH * 4, hipMemcpyDeviceToDevice, stream);

    // Pack weights to WMMA B-fragment bf16 layout
    k_pack_w<<<(G3 * HH + 255) / 256, 256, 0, stream>>>(W_hh_p, G3, HH, HH, whh_p_pk);
    k_pack_w<<<(HH * HH + 255) / 256, 256, 0, stream>>>(W_fc_p, HH, HH, HH, wfc_p_pk);
    k_pack_w<<<(G3 * HH + 255) / 256, 256, 0, stream>>>(W_ih_a, G3, HH, 259, wihf_pk);
    k_pack_w<<<(G3 * HH + 255) / 256, 256, 0, stream>>>(W_hh_a, G3, HH, HH, whh_a_pk);
    k_pack_w<<<(HH * HH + 255) / 256, 256, 0, stream>>>(W_fc_a, HH, HH, HH, wfc_a_pk);
    k_pack_w<<<(HH * HH + 255) / 256, 256, 0, stream>>>(W_m1,   HH, HH, HH, wm1_pk);

    // Step-invariant input-GEMM precompute
    {
        size_t n1 = (size_t)NBAE * G3;
        k_gi_p<<<(unsigned)((n1 + 255) / 256), 256, 0, stream>>>(obs, phys, W_ih_p, b_ih_p, gi_p);
        size_t n2 = (size_t)NBA * G3;
        k_gia_goal<<<(unsigned)((n2 + 255) / 256), 256, 0, stream>>>(goals, W_ih_a, b_ih_a, gia_goal);
    }

    // Sequential T-step recurrence
    for (int tstep = 0; tstep < NT; ++tstep) {
        k_phys_step<<<NBAE / 128, 256, 0, stream>>>(gi_p, b_hh_p, b_fc_p,
                                                    whh_p_pk, wfc_p_pk, mp, proc);
        k_pool<<<NBA, HH, 0, stream>>>(proc, feat);
        k_act_step<<<NBA / 128, 256, 0, stream>>>(feat, gia_goal, b_hh_a, b_fc_a, b_m1,
                                                  W_m2, b_m2,
                                                  wihf_pk, whh_a_pk, wfc_a_pk, wm1_pk,
                                                  ma, out + (size_t)tstep * NBA * 2);
    }
}
